// Molecule3DBias_54236847014173
// MI455X (gfx1250) — matrix operand order repaired and checked by reading, not compiled, not run on hardware
//
#include <hip/hip_runtime.h>
#include <hip/hip_bf16.h>

typedef __attribute__((ext_vector_type(16))) _Float16 v16h;
typedef __attribute__((ext_vector_type(8)))  _Float16 v8h;
typedef __attribute__((ext_vector_type(8)))  float    v8f;

#define BDIM 256
#define NN_  256   // N
#define KK   128   // num_kernel
#define HH   32    // heads
#define DD   768   // embed dim

// 16-bit A-matrix 16x32 lane/element -> K mapping (CDNA5 ISA 7.12.2):
// lanes 0-15: v0..3 -> K=2v+h, v4..7 -> K=16+2(v-4)+h ; lanes 16-31: +8
__device__ __forceinline__ int a_kpat(int e, int lane) {
  int v = e >> 1, h = e & 1;
  return ((v >> 2) << 4) + ((lane >> 4) << 3) + ((v & 3) << 1) + h;
}

__global__ __launch_bounds__(BDIM)
void molecule3d_fused(const float* __restrict__ pos,     // [B,N,3]
                      const int*   __restrict__ xin,     // [B,N,9]
                      const int*   __restrict__ nte,     // [B,N,N,2]
                      const float* __restrict__ means,   // [K]
                      const float* __restrict__ stds,    // [K]
                      const float* __restrict__ mul_w,   // [E]
                      const float* __restrict__ bias_w,  // [E]
                      const float* __restrict__ w1,      // [K,K]
                      const float* __restrict__ b1,      // [K]
                      const float* __restrict__ w2,      // [K,H]
                      const float* __restrict__ b2,      // [H]
                      const float* __restrict__ edge_w,  // [K,D]
                      const float* __restrict__ edge_b,  // [D]
                      float* __restrict__ attn,          // [B,H,N,N]
                      float* __restrict__ merge,         // [B,N,D]
                      float* __restrict__ delta)         // [B,N,N,3]
{
  __shared__ _Float16 s_w1t[KK * KK];    // transposed: [n][k]  32 KB
  __shared__ _Float16 s_w2t[HH * KK];    // transposed: [n][k]   8 KB
  __shared__ _Float16 s_hid[8 * 16 * 32];// per-wave staging     8 KB
  __shared__ float    s_sum[16 * KK];    // masked K-sums        8 KB
  __shared__ float    s_pos[NN_ * 3];
  __shared__ float    s_mask[NN_];
  __shared__ float    s_mean[KK], s_inv[KK], s_norm[KK];
  __shared__ float    s_b1[KK];
  __shared__ float    s_b2[HH];

  const int tid  = threadIdx.x;
  const int lane = tid & 31;
  const int wave = tid >> 5;
  const int b    = blockIdx.x >> 4;
  const int i0   = (blockIdx.x & 15) << 4;
  const int m    = lane & 15;          // row within 16-tile
  const int lg   = lane >> 4;          // lane group 0/1

  // ---------------- init LDS ----------------
  for (int idx = tid; idx < KK * KK; idx += BDIM) {
    int n = idx >> 7, k = idx & 127;
    s_w1t[idx] = (_Float16)w1[k * KK + n];
  }
  for (int idx = tid; idx < HH * KK; idx += BDIM) {
    int n = idx >> 7, k = idx & 127;
    s_w2t[idx] = (_Float16)w2[k * HH + n];
  }
  for (int idx = tid; idx < NN_ * 3; idx += BDIM)
    s_pos[idx] = pos[(size_t)(b * NN_) * 3 + idx];
  if (tid < NN_) {
    int allz = 1;
    for (int c = 0; c < 9; ++c) allz &= (xin[(size_t)(b * NN_ + tid) * 9 + c] == 0);
    s_mask[tid] = allz ? 1.0f : 0.0f;
  }
  if (tid < KK) {
    s_mean[tid] = means[tid];
    float s = fabsf(stds[tid]) + 0.01f;
    s_inv[tid]  = 1.0f / s;
    s_norm[tid] = 1.0f / (2.50662627f * s);   // sqrt(2*3.14159)*std
    s_b1[tid]   = b1[tid];
  }
  if (tid < HH) s_b2[tid] = b2[tid];
  for (int idx = tid; idx < 16 * KK; idx += BDIM) s_sum[idx] = 0.0f;
  __syncthreads();

  // ---------------- delta_pos ----------------
  for (int p = tid; p < 16 * NN_; p += BDIM) {
    int il = p >> 8, j = p & 255;
    int i = i0 + il;
    float dx = s_pos[j * 3 + 0] - s_pos[i * 3 + 0];
    float dy = s_pos[j * 3 + 1] - s_pos[i * 3 + 1];
    float dz = s_pos[j * 3 + 2] - s_pos[i * 3 + 2];
    float dist = sqrtf(dx * dx + dy * dy + dz * dz + 1e-12f);
    float inv = 1.0f / (dist + 1e-5f);
    size_t off = ((size_t)((b * NN_ + i) * NN_ + j)) * 3;
    delta[off + 0] = dx * inv;
    delta[off + 1] = dy * inv;
    delta[off + 2] = dz * inv;
  }

  const v8f vzero = {0.f, 0.f, 0.f, 0.f, 0.f, 0.f, 0.f, 0.f};
  _Float16* hptr = &s_hid[wave * 512];   // this wave's 16x32 staging

  // ---------------- main fused pipeline ----------------
  for (int ipass = 0; ipass < 2; ++ipass) {
    const int il = wave * 2 + ipass;     // local i row 0..15
    const int i  = i0 + il;
    float acc[64];
    #pragma unroll
    for (int q = 0; q < 64; ++q) acc[q] = 0.0f;

    for (int jb = 0; jb < 16; ++jb) {
      const int jbase = jb << 4;
      const int j = jbase + m;
      // per-pair scalars (lanes m and m+16 duplicate, by design)
      float dx = s_pos[j * 3 + 0] - s_pos[i * 3 + 0];
      float dy = s_pos[j * 3 + 1] - s_pos[i * 3 + 1];
      float dz = s_pos[j * 3 + 2] - s_pos[i * 3 + 2];
      float dist = sqrtf(dx * dx + dy * dy + dz * dz + 1e-12f);
      size_t poff = ((size_t)((b * NN_ + i) * NN_ + j)) * 2;
      int t0 = nte[poff + 0], t1 = nte[poff + 1];
      float mulv = mul_w[t0] + mul_w[t1];
      float biav = bias_w[t0] + bias_w[t1];
      float xg = mulv * dist + biav;
      float wkeep = 1.0f - s_mask[j];

      // Gaussian features straight into WMMA A-fragments (registers only)
      v16h a1[4];
      #pragma unroll
      for (int ks = 0; ks < 4; ++ks) {
        #pragma unroll
        for (int e = 0; e < 16; ++e) {
          int k = ks * 32 + a_kpat(e, lane);
          float t = (xg - s_mean[k]) * s_inv[k];
          float g = __expf(-0.5f * t * t) * s_norm[k];
          a1[ks][e] = (_Float16)g;
          acc[ks * 16 + e] += g * wkeep;
        }
      }

      // ---- GEMM1: [16x128] @ w1[128x128] ----
      v8f c[8];
      #pragma unroll
      for (int nb = 0; nb < 8; ++nb) c[nb] = vzero;
      #pragma unroll
      for (int ks = 0; ks < 4; ++ks) {
        #pragma unroll
        for (int nb = 0; nb < 8; ++nb) {
          union { v16h v; v8h h[2]; } bw;
          int off = (nb * 16 + m) * KK + ks * 32 + (lg << 3);  // table K order
          bw.h[0] = *(const v8h*)&s_w1t[off];
          bw.h[1] = *(const v8h*)&s_w1t[off + 16];
          c[nb] = __builtin_amdgcn_wmma_f32_16x16x32_f16(
              false, a1[ks], false, bw.v, (short)0, c[nb], false, false);
        }
      }

      // ---- GELU + lane-transpose via LDS + GEMM2: [16x128] @ w2[128x32] ----
      v8f c2[2];
      c2[0] = vzero; c2[1] = vzero;
      #pragma unroll
      for (int ks2 = 0; ks2 < 4; ++ks2) {
        #pragma unroll
        for (int nl = 0; nl < 2; ++nl) {
          int nb = ks2 * 2 + nl;
          float bb1 = s_b1[nb * 16 + m];
          #pragma unroll
          for (int r = 0; r < 8; ++r) {
            float xv = c[nb][r] + bb1;
            float gl = 0.5f * xv * (1.0f + erff(xv * 0.70710678f));
            int mm = r + (lg << 3);
            hptr[mm * 32 + nl * 16 + m] = (_Float16)gl;
          }
        }
        asm volatile("s_wait_dscnt 0" ::: "memory");  // stores visible before reads
        union { v16h v; v8h h[2]; } a2;
        int ho = m * 32 + (lg << 3);
        a2.h[0] = *(const v8h*)&hptr[ho];
        a2.h[1] = *(const v8h*)&hptr[ho + 16];
        #pragma unroll
        for (int nb2 = 0; nb2 < 2; ++nb2) {
          union { v16h v; v8h h[2]; } bw;
          int off = (nb2 * 16 + m) * KK + ks2 * 32 + (lg << 3);
          bw.h[0] = *(const v8h*)&s_w2t[off];
          bw.h[1] = *(const v8h*)&s_w2t[off + 16];
          c2[nb2] = __builtin_amdgcn_wmma_f32_16x16x32_f16(
              false, a2.v, false, bw.v, (short)0, c2[nb2], false, false);
        }
      }

      // ---- write attn bias [b,h,i,j] with -inf mask on j ----
      #pragma unroll
      for (int nb2 = 0; nb2 < 2; ++nb2) {
        int h = nb2 * 16 + m;
        float bb = s_b2[h];
        #pragma unroll
        for (int r = 0; r < 8; ++r) {
          int jj = jbase + r + (lg << 3);
          float val = c2[nb2][r] + bb;
          if (s_mask[jj] != 0.0f) val = -__builtin_huge_valf();
          attn[(((size_t)(b * HH + h) * NN_ + i)) * NN_ + jj] = val;
        }
      }
    } // jb

    // flush masked K-sum for this i row (lanes of a half-wave share k slots)
    #pragma unroll
    for (int ks = 0; ks < 4; ++ks) {
      #pragma unroll
      for (int e = 0; e < 16; ++e)
        atomicAdd(&s_sum[il * KK + ks * 32 + a_kpat(e, lane)], acc[ks * 16 + e]);
    }
  } // ipass

  __syncthreads();

  // ---------------- merge: [16x128] @ edge_w[128x768] + edge_b ----------------
  {
    v16h as[4];
    #pragma unroll
    for (int ks = 0; ks < 4; ++ks) {
      #pragma unroll
      for (int e = 0; e < 16; ++e)
        as[ks][e] = (_Float16)s_sum[m * KK + ks * 32 + a_kpat(e, lane)];
    }
    for (int ct = wave; ct < DD / 16; ct += 8) {
      int col = ct * 16 + m;
      v8f cm = vzero;
      #pragma unroll
      for (int ks = 0; ks < 4; ++ks) {
        v16h bw;
        #pragma unroll
        for (int e = 0; e < 16; ++e) {
          int k = ks * 32 + (lg << 3) + (e & 7) + ((e >> 3) << 4);   // B-layout: table K order
          bw[e] = (_Float16)edge_w[(size_t)k * DD + col];
        }
        cm = __builtin_amdgcn_wmma_f32_16x16x32_f16(
            false, as[ks], false, bw, (short)0, cm, false, false);
      }
      float eb = edge_b[col];
      #pragma unroll
      for (int r = 0; r < 8; ++r) {
        int row = i0 + r + (lg << 3);
        merge[((size_t)(b * NN_ + row)) * DD + col] = cm[r] + eb;
      }
    }
  }
}

extern "C" void kernel_launch(void* const* d_in, const int* in_sizes, int n_in,
                              void* d_out, int out_size, void* d_ws, size_t ws_size,
                              hipStream_t stream) {
  const float* pos    = (const float*)d_in[0];
  const int*   x      = (const int*)  d_in[1];
  const int*   nte    = (const int*)  d_in[2];
  const float* means  = (const float*)d_in[3];
  const float* stds   = (const float*)d_in[4];
  const float* mul_w  = (const float*)d_in[5];
  const float* bias_w = (const float*)d_in[6];
  const float* w1     = (const float*)d_in[7];
  const float* b1     = (const float*)d_in[8];
  const float* w2     = (const float*)d_in[9];
  const float* b2     = (const float*)d_in[10];
  const float* edge_w = (const float*)d_in[11];
  const float* edge_b = (const float*)d_in[12];

  float* out   = (float*)d_out;
  float* attn  = out;                                   // [8,32,256,256]
  float* merge = attn + (size_t)8 * 32 * 256 * 256;     // [8,256,768]
  float* delta = merge + (size_t)8 * 256 * 768;         // [8,256,256,3]

  dim3 grid(8 * (256 / 16));  // 128 workgroups: (b, i-tile)
  dim3 block(BDIM);           // 8 waves (wave32)
  molecule3d_fused<<<grid, block, 0, stream>>>(pos, x, nte, means, stds, mul_w,
                                               bias_w, w1, b1, w2, b2, edge_w,
                                               edge_b, attn, merge, delta);
}
